// HConvolution2D_88055419503163
// MI455X (gfx1250) — compile-verified, matrix operand order checked by reading, not run on hardware
//
#include <hip/hip_runtime.h>
#include <cmath>

typedef __attribute__((ext_vector_type(16))) __bf16 v16bf;
typedef __attribute__((ext_vector_type(8)))  float  v8f;
typedef unsigned int uint32x4 __attribute__((ext_vector_type(4)));
typedef int          int32x8  __attribute__((ext_vector_type(8)));
typedef int          int32x4  __attribute__((ext_vector_type(4)));

static __device__ __forceinline__ unsigned short f2bf(float f) {
    unsigned int u = __float_as_uint(f);
    unsigned int r = ((u >> 16) & 1u) + 0x7FFFu;   // round-to-nearest-even
    return (unsigned short)((u + r) >> 16);
}

// ---------------------------------------------------------------------------
// Stage A: logmap0 over channel axis; store u*beta as bf16 NHWC + q = ||v||^2
// x: [32,64,64,64] NCHW f32.  u: [32,64,64,64] NHWC bf16.  q: [32,64,64] f32
// ---------------------------------------------------------------------------
__global__ __launch_bounds__(256) void logmap_kernel(
    const float* __restrict__ x, unsigned short* __restrict__ u,
    float* __restrict__ q, float beta) {
    int pix = blockIdx.x * 256 + threadIdx.x;         // 131072 pixels
    int b = pix >> 12, hw = pix & 4095;
    const float* xp = x + ((size_t)b << 18) + hw;     // channel stride 4096
    float s2 = 0.f;
    for (int c = 0; c < 64; ++c) { float v = xp[c << 12]; s2 += v * v; }
    float n  = sqrtf(s2);
    float sn = fminf(fmaxf(n, 1e-15f), 1.f - 1e-7f);
    float f  = atanhf(sn) / fmaxf(n, 1e-15f) * beta;  // fold beta_ratio in
    float qs = 0.f;
    unsigned short* up = u + (size_t)pix * 64;
    for (int c = 0; c < 64; ++c) {
        float v = f * xp[c << 12];
        qs += v * v;
        up[c] = f2bf(v);
    }
    q[pix] = qs;
}

// ---------------------------------------------------------------------------
// Stage B1: column norms of weights [576,128] + sinh/cosh(2*bias)
// ---------------------------------------------------------------------------
__global__ void wprep_kernel(const float* __restrict__ w,
                             const float* __restrict__ bias,
                             float* __restrict__ znorm,
                             float* __restrict__ sinhd,
                             float* __restrict__ coshd) {
    int o = threadIdx.x;                               // 128 threads
    float s2 = 0.f;
    for (int d = 0; d < 576; ++d) { float v = w[d * 128 + o]; s2 += v * v; }
    znorm[o] = sqrtf(s2);
    float drcr = 2.f * bias[o];
    sinhd[o] = sinhf(drcr);
    coshd[o] = coshf(drcr);
}

// ---------------------------------------------------------------------------
// Stage B2: pack z_unit into per-lane WMMA-B bf16 fragment layout.
// Bpk index = ((kc*8 + nt)*32 + lane)*16 + j
//   lane<16: N=lane, koff=0 ; lane>=16: koff=8
//   element j -> k_local = koff + (j<8 ? j : j+8)   (K pairs: 0..7,16..23)
// jax unfold order: d_ref = ch*9 + pos ; ours: k = pos*64 + ch
// ---------------------------------------------------------------------------
__global__ __launch_bounds__(256) void wpack_kernel(
    const float* __restrict__ w, const float* __restrict__ znorm,
    unsigned short* __restrict__ Bpk) {
    int idx = blockIdx.x * 256 + threadIdx.x;          // 73728 elements
    int kc = idx >> 12;
    int r1 = idx & 4095;
    int nt = r1 >> 9;
    int r2 = r1 & 511;
    int L  = r2 >> 4;
    int j  = r2 & 15;
    int koff   = (L >= 16) ? 8 : 0;
    int klocal = koff + ((j < 8) ? j : j + 8);
    int k   = kc * 32 + klocal;
    int pos = k >> 6, ch = k & 63;
    int n   = nt * 16 + (L & 15);
    int dref = ch * 9 + pos;
    float zn = fmaxf(znorm[n], 1e-15f);
    Bpk[idx] = f2bf(w[dref * 128 + n] / zn);
}

// ---------------------------------------------------------------------------
// Stage C: implicit-GEMM WMMA + hyperbolic epilogue.
// Block: 256 thr (8 waves). Tile: 128 pixels (2 image rows) x 128 outputs.
// B (144 KB) staged to LDS by the Tensor Data Mover (one DMA, TENSORcnt).
// ---------------------------------------------------------------------------
__global__ __launch_bounds__(256) void hconv_gemm_kernel(
    const unsigned short* __restrict__ u, const float* __restrict__ q,
    const unsigned short* __restrict__ Bpk,
    const float* __restrict__ znorm, const float* __restrict__ sinhd,
    const float* __restrict__ coshd, float* __restrict__ out) {
    extern __shared__ char smem[];
    unsigned short* sB = (unsigned short*)smem;        // 147456 B, aliased as yT later
    float* sArr   = (float*)(smem + 147456);           // per-pixel tanh scale
    float* cx2Arr = sArr + 128;                        // per-pixel cx2

    int tid  = threadIdx.x;
    int wave = tid >> 5, lane = tid & 31;
    int pixBase = blockIdx.x << 7;                     // 1024 blocks
    int b  = pixBase >> 12;
    int h0 = (pixBase >> 6) & 63;                      // rows h0, h0+1

    __builtin_prefetch(u + (size_t)pixBase * 64, 0, 3);

#if __has_builtin(__builtin_amdgcn_tensor_load_to_lds)
    // ---- TDM: one DMA moves all 147456 B of packed B into LDS offset 0 ----
    if (tid < 32) {
        unsigned long long ga = (unsigned long long)(size_t)Bpk;
        uint32x4 g0;
        g0.x = 1u;                                  // count=1, user descriptor
        g0.y = 0u;                                  // lds_addr = 0 (dyn LDS base)
        g0.z = (unsigned int)ga;                    // global_addr[31:0]
        g0.w = (unsigned int)(ga >> 32) | (2u << 30); // addr[56:32], type=2
        int32x8 g1;
        g1[0] = 0x20000;                            // data_size = 4 bytes
        g1[1] = (int)0x90000000;                    // tensor_dim0 = 36864 (lo16 in [31:16])
        g1[2] = 0x00010000;                         // dim0 hi = 0, tensor_dim1 = 1
        g1[3] = (int)0x90000000;                    // dim1 hi = 0, tile_dim0 = 36864
        g1[4] = 0;                                  // tile_dim1 = tile_dim2 = 0
        g1[5] = 36864;                              // tensor_dim0_stride (lo32)
        g1[6] = 0;                                  // stride hi, dim1_stride lo
        g1[7] = 0;
        int32x4 gz4 = {0, 0, 0, 0};                 // groups 2/3 unused (<=2D)
        int32x8 gz8 = {0, 0, 0, 0, 0, 0, 0, 0};
        __builtin_amdgcn_tensor_load_to_lds(g0, g1, gz4, gz4, gz8, 0);
    }
#if __has_builtin(__builtin_amdgcn_s_wait_tensorcnt)
    __builtin_amdgcn_s_wait_tensorcnt(0);
#else
    asm volatile("s_wait_tensorcnt 0x0" ::: "memory");
#endif
#else
    // ---- Fallback: cooperative copy through VGPRs ----
    {
        const uint4* gB = (const uint4*)Bpk;
        uint4* lB = (uint4*)sB;
        for (int i = tid; i < 9216; i += 256) lB[i] = gB[i];
    }
#endif

    // Per-pixel hyperbolic scalars: n2 = sum of q over 3x3 neighborhood.
    if (tid < 128) {
        int h = h0 + (tid >> 6), w = tid & 63;
        float n2 = 0.f;
        for (int dy = -1; dy <= 1; ++dy)
            for (int dx = -1; dx <= 1; ++dx) {
                int hy = h + dy, wx = w + dx;
                if (hy >= 0 && hy < 64 && wx >= 0 && wx < 64)
                    n2 += q[(b << 12) + (hy << 6) + wx];
            }
        float ne = fmaxf(sqrtf(n2), 1e-15f);
        float s  = tanhf(ne) / ne;                     // expmap0 scale
        sArr[tid]   = s;
        cx2Arr[tid] = s * s * n2;                      // ||rc*xu||^2
    }
    __syncthreads();

    int prl = (wave << 4) + (lane & 15);               // this wave's 16 pixels
    int hpx = h0 + (prl >> 6), wpx = prl & 63;
    int koff = (lane >= 16) ? 8 : 0;

    v8f zero = {};
    v8f acc[8];
#pragma unroll
    for (int t = 0; t < 8; ++t) acc[t] = zero;

    union AU { v16bf v; uint4 qv[2]; };

    for (int pos = 0; pos < 9; ++pos) {
        int dy = pos / 3 - 1, dx = pos % 3 - 1;
        int hy = hpx + dy, wx = wpx + dx;
        bool valid = (hy >= 0) && (hy < 64) && (wx >= 0) && (wx < 64);
#pragma unroll
        for (int cc = 0; cc < 64; cc += 32) {
            AU a;
            if (valid) {
                const uint4* p = (const uint4*)(u +
                    (((size_t)((b << 12) + (hy << 6) + wx)) << 6) + cc + koff);
                a.qv[0] = p[0];                        // K koff..koff+7
                a.qv[1] = p[2];                        // K koff+16..koff+23
            } else {
                a.qv[0] = make_uint4(0, 0, 0, 0);
                a.qv[1] = make_uint4(0, 0, 0, 0);
            }
            int kcIdx = pos * 2 + (cc >> 5);
#pragma unroll
            for (int t = 0; t < 8; ++t) {
                v16bf bb = *(const v16bf*)(sB + (((kcIdx * 8 + t) * 32 + lane) << 4));
                acc[t] = __builtin_amdgcn_wmma_f32_16x16x32_bf16(
                    false, a.v, false, bb, (short)0, acc[t], false, false);
            }
        }
    }
    __syncthreads();                                   // sB no longer needed

    // Epilogue: mlr -> sinh -> ball projection; write transposed to LDS.
    float* yT = (float*)smem;                          // [128][129] padded
    float zn[8], shd[8], chd[8];
    int ob = lane & 15;
#pragma unroll
    for (int t = 0; t < 8; ++t) {
        int o = t * 16 + ob;
        zn[t] = znorm[o]; shd[t] = sinhd[o]; chd[t] = coshd[o];
    }
    int hi = lane >> 4;                                // 0: M=r ; 1: M=r+8
#pragma unroll
    for (int r = 0; r < 8; ++r) {
        int pr = (wave << 4) + r + (hi << 3);
        float s = sArr[pr], cx2 = cx2Arr[pr];
        float invden = 1.f / fmaxf(1.f - cx2, 1e-15f);
        float yv[8];
        float s2 = 0.f;
#pragma unroll
        for (int t = 0; t < 8; ++t) {
            float G   = acc[t][r];
            float num = 2.f * s * G * chd[t] - (1.f + cx2) * shd[t];
            float mlr = 2.f * zn[t] * asinhf(num * invden);
            float y   = sinhf(mlr);
            yv[t] = y;
            s2 += y * y;
        }
        // reduce over the 16 lanes sharing this pixel (half-wave groups)
        s2 += __shfl_xor(s2, 1, 32);
        s2 += __shfl_xor(s2, 2, 32);
        s2 += __shfl_xor(s2, 4, 32);
        s2 += __shfl_xor(s2, 8, 32);
        float dn = 1.f / (1.f + sqrtf(1.f + s2));
#pragma unroll
        for (int t = 0; t < 8; ++t)
            yT[pr * 129 + t * 16 + ob] = yv[t] * dn;
    }
    __syncthreads();

    // Coalesced NCHW store: out[b][o][h][w]
    for (int e = tid; e < 16384; e += 256) {
        int o = e >> 7, pr = e & 127;
        int hh = h0 + (pr >> 6), ww = pr & 63;
        out[(((size_t)(b * 128 + o)) << 12) + (hh << 6) + ww] = yT[pr * 129 + o];
    }
}

// ---------------------------------------------------------------------------
extern "C" void kernel_launch(void* const* d_in, const int* in_sizes, int n_in,
                              void* d_out, int out_size, void* d_ws, size_t ws_size,
                              hipStream_t stream) {
    const float* x    = (const float*)d_in[0];   // [32,64,64,64]
    const float* wts  = (const float*)d_in[1];   // [576,128]
    const float* bias = (const float*)d_in[2];   // [128]
    float* out = (float*)d_out;

    char* ws = (char*)d_ws;
    unsigned short* u   = (unsigned short*)ws;                       // 33554432 B
    float*          q   = (float*)(ws + 33554432);                   //   524288 B
    unsigned short* Bpk = (unsigned short*)(ws + 33554432 + 524288); //   147456 B
    float* znorm = (float*)(ws + 33554432 + 524288 + 147456);        // 128 f
    float* sinhd = znorm + 128;
    float* coshd = znorm + 256;

    // beta_ratio = B(288,0.5)/B(32,0.5); Gamma(0.5) cancels.
    double br = exp((::lgamma(288.0) - ::lgamma(288.5)) -
                    (::lgamma(32.0)  - ::lgamma(32.5)));

    logmap_kernel<<<512, 256, 0, stream>>>(x, u, q, (float)br);
    wprep_kernel<<<1, 128, 0, stream>>>(wts, bias, znorm, sinhd, coshd);
    wpack_kernel<<<288, 256, 0, stream>>>(wts, znorm, Bpk);
    hconv_gemm_kernel<<<1024, 256, 148480, stream>>>(u, q, Bpk, znorm, sinhd,
                                                     coshd, out);
}
